// MultiHeadedAttention_8031588843642
// MI455X (gfx1250) — matrix operand order repaired and checked by reading, not compiled, not run on hardware
//
#include <hip/hip_runtime.h>
#include <hip/hip_bf16.h>

// ---------------- Problem constants (from reference) ----------------
#define BB 8
#define SS 1024
#define DD 768
#define HH 16
#define PDIM 48          // per-head dim
#define MTOT (BB * SS)   // 8192 rows

// ---------------- Vector types ----------------
typedef __attribute__((ext_vector_type(16))) _Float16 f16x16;
typedef __attribute__((ext_vector_type(8)))  _Float16 f16x8;
typedef __attribute__((ext_vector_type(4)))  _Float16 f16x4;
typedef __attribute__((ext_vector_type(8)))  float    f32x8;
typedef __attribute__((ext_vector_type(4)))  float    f32x4;

// ---------------- WMMA wrapper ----------------
__device__ __forceinline__ f32x8 wmma_f16(f16x16 a, f16x16 b, f32x8 c) {
    // D(f32 16x16) = A(f16 16x32) * B(f16 32x16) + C
    return __builtin_amdgcn_wmma_f32_16x16x32_f16(
        /*neg_a=*/false, a, /*neg_b=*/false, b,
        /*c_mod=*/(short)0, c, /*reuse_a=*/false, /*reuse_b=*/false);
}

// ---------------- CDNA5 async global->LDS copy (ASYNCcnt path) ----------------
// lds_off: byte offset within the wave's LDS allocation (dynamic smem @ 0 here).
__device__ __forceinline__ void async_cp_b128(unsigned lds_off, const void* gaddr) {
    asm volatile("global_load_async_to_lds_b128 %0, %1, off"
                 :: "v"(lds_off), "v"(gaddr) : "memory");
}
__device__ __forceinline__ void wait_async0() {
    asm volatile("s_wait_asynccnt 0x0" ::: "memory");
}

// A fragment (f16 source in LDS, row-major, 16 rows x 32 cols tile).
// Lane L: m = L&15, hf = L>>4. K runs: {hf*8..+7, 16+hf*8..+7}.
__device__ __forceinline__ f16x16 load_frag_a_f16(const _Float16* base, int stride, int lane) {
    int m = lane & 15, hf = lane >> 4;
    const _Float16* p = base + m * stride;
    f16x8 lo = *(const f16x8*)(p + hf * 8);
    f16x8 hi = *(const f16x8*)(p + 16 + hf * 8);
    f16x16 a;
#pragma unroll
    for (int i = 0; i < 8; ++i) { a[i] = lo[i]; a[8 + i] = hi[i]; }
    return a;
}

// A fragment from f32 LDS (convert to f16 in registers).
__device__ __forceinline__ f16x16 load_frag_a_f32(const float* base, int stride, int lane) {
    int m = lane & 15, hf = lane >> 4;
    const float* p = base + m * stride;
    f32x4 x0 = *(const f32x4*)(p + hf * 8);
    f32x4 x1 = *(const f32x4*)(p + hf * 8 + 4);
    f32x4 x2 = *(const f32x4*)(p + 16 + hf * 8);
    f32x4 x3 = *(const f32x4*)(p + 16 + hf * 8 + 4);
    f16x16 a;
#pragma unroll
    for (int i = 0; i < 4; ++i) {
        a[i]      = (_Float16)x0[i];
        a[4 + i]  = (_Float16)x1[i];
        a[8 + i]  = (_Float16)x2[i];
        a[12 + i] = (_Float16)x3[i];
    }
    return a;
}

// B fragment. LDS holds B transposed: Bt[n][k], `stride` halfs between n-rows.
// Lane L: n = L&15, hf = L>>4. K runs: {hf*8..+7, 16+hf*8..+7}.
__device__ __forceinline__ f16x16 load_frag_b(const _Float16* baseT, int stride, int lane) {
    int n = lane & 15, hf = lane >> 4;
    const _Float16* p = baseT + n * stride;
    f16x8 lo = *(const f16x8*)(p + hf * 8);
    f16x8 hi = *(const f16x8*)(p + 16 + hf * 8);
    f16x16 b;
#pragma unroll
    for (int i = 0; i < 8; ++i) { b[i] = lo[i]; b[8 + i] = hi[i]; }
    return b;
}

// 4-wide load + convert to f16 (overloads for f32 / f16 A operand)
__device__ __forceinline__ f16x4 ld4_cvt(const float* p) {
    f32x4 v = *(const f32x4*)p;
    f16x4 r;
#pragma unroll
    for (int i = 0; i < 4; ++i) r[i] = (_Float16)v[i];
    return r;
}
__device__ __forceinline__ f16x4 ld4_cvt(const _Float16* p) {
    return *(const f16x4*)p;
}

// =====================================================================
// GEMM: C[M,N] = A[M,K] * W[K,N] + bias[N]
// 256 threads (8 waves), BM=BN=128, BK=32. Waves in 4x2 grid; each wave
// owns 2x4 output tiles -> 6 fragment loads feed 8 WMMAs per K-step.
// =====================================================================
template <typename TA, typename TO>
__global__ __launch_bounds__(256)
void gemm_wmma_kernel(const TA* __restrict__ A, const float* __restrict__ W,
                      const float* __restrict__ bias, TO* __restrict__ C,
                      int M, int K, int N) {
    __shared__ __align__(16) _Float16 sA[128 * 32];   // [m][k]
    __shared__ __align__(16) _Float16 sB[128 * 32];   // Bt[n][k]

    const int tid  = threadIdx.x;
    const int wid  = tid >> 5;
    const int lane = tid & 31;
    const int m0 = blockIdx.y * 128;
    const int n0 = blockIdx.x * 128;
    const int wr = wid & 3;      // wave tile-row group (2 tile rows each)
    const int wc = wid >> 2;     // wave tile-col group (4 tile cols each)

    f32x8 acc[2][4] = {};

    for (int k0 = 0; k0 < K; k0 += 32) {
        __syncthreads();
        // stage A: 128x32 f16, 4-wide chunks (16 elems/thread)
        // stage B: transposed Bt[n][k]; coalesced f32x4 global reads
#pragma unroll
        for (int t = 0; t < 4; ++t) {
            int cidx = tid + t * 256;                  // 1024 chunks of 4
            int ar = cidx >> 3, ac = (cidx & 7) * 4;
            *(f16x4*)&sA[ar * 32 + ac] = ld4_cvt(&A[(size_t)(m0 + ar) * K + k0 + ac]);
            int kk = cidx >> 5, nb = (cidx & 31) * 4;
            f32x4 wv = *(const f32x4*)&W[(size_t)(k0 + kk) * N + n0 + nb];
#pragma unroll
            for (int i = 0; i < 4; ++i) sB[(nb + i) * 32 + kk] = (_Float16)wv[i];
        }
        if (k0 + 32 < K) {       // prefetch next K-panel
            __builtin_prefetch(&A[(size_t)(m0 + (tid >> 1)) * K + k0 + 32], 0, 1);
            __builtin_prefetch(&W[(size_t)(k0 + 32 + (tid >> 3)) * N + n0 + (tid & 7) * 16], 0, 1);
        }
        __syncthreads();

        f16x16 af[2], bf[4];
#pragma unroll
        for (int i = 0; i < 2; ++i)
            af[i] = load_frag_a_f16(sA + ((wr * 2 + i) * 16) * 32, 32, lane);
#pragma unroll
        for (int c = 0; c < 4; ++c)
            bf[c] = load_frag_b(sB + ((wc * 4 + c) * 16) * 32, 32, lane);
#pragma unroll
        for (int i = 0; i < 2; ++i)
#pragma unroll
            for (int c = 0; c < 4; ++c)
                acc[i][c] = wmma_f16(af[i], bf[c], acc[i][c]);
    }

    const int hf = lane >> 4, nl = lane & 15;
#pragma unroll
    for (int i = 0; i < 2; ++i) {
#pragma unroll
        for (int c = 0; c < 4; ++c) {
            int col = n0 + (wc * 4 + c) * 16 + nl;
            float bi = bias[col];
#pragma unroll
            for (int r = 0; r < 8; ++r) {
                int row = m0 + (wr * 2 + i) * 16 + r + hf * 8;
                C[(size_t)row * N + col] = (TO)(acc[i][c][r] + bi);
            }
        }
    }
}

// =====================================================================
// Attention LDS layout (dynamic smem, byte offsets within LDS alloc)
// =====================================================================
#define SS_BYTES  (64 * 1024 * 4)          // f32 score strip [64][1024]
#define SQ_OFF    SS_BYTES                 // f16 Q tile [64][64]
#define SKV_OFF   (SQ_OFF + 64 * 64 * 2)   // f16 K/V tile [64][64]
#define ATTN_LDS_BYTES (SKV_OFF + 64 * 64 * 2)

// Stage 64 rows x 48 halfs (row stride DD) -> LDS [64][64] via CDNA5 async
// global->LDS b128 copies; pad cols 48..63 zeroed with plain ds stores.
// Round 0: 256 unconditional copy tasks. Round 1 (wave-uniform split):
// waves 0..3 copy tasks 256..383, waves 4..7 write the 128 pad vectors.
__device__ __forceinline__ void stage_rows48_async(const _Float16* __restrict__ src,
                                                   unsigned lds_base,
                                                   _Float16* __restrict__ dst_pad,
                                                   int tid) {
    {
        int row = tid / 6, ch = tid - row * 6;
        async_cp_b128(lds_base + (unsigned)(row * 64 + ch * 8) * 2,
                      src + (size_t)row * DD + ch * 8);
    }
    if (tid < 128) {
        int tk = tid + 256;
        int row = tk / 6, ch = tk - row * 6;
        async_cp_b128(lds_base + (unsigned)(row * 64 + ch * 8) * 2,
                      src + (size_t)row * DD + ch * 8);
    } else {
        int tk2 = tid - 128;
        int row = tk2 >> 1, ch = tk2 & 1;
        f16x8 z = {};
        *(f16x8*)(dst_pad + row * 64 + 48 + ch * 8) = z;
    }
}

// =====================================================================
// Attention: one workgroup per (b, h, 64-query strip).
// Full 64x1024 f32 score strip in LDS (CDNA5 320KB/WGP) -> exact
// single-pass softmax. Phase 1: S=QK^T (512 wmma/WG); Phase 2: O=PV
// (384 wmma/WG) with P converted f32->f16 at fragment load.
// =====================================================================
__global__ __launch_bounds__(256)
void attn_wmma_kernel(const _Float16* __restrict__ Qg,
                      const _Float16* __restrict__ Kg,
                      const _Float16* __restrict__ Vg,
                      _Float16* __restrict__ Og) {
    extern __shared__ __align__(16) char smem[];
    float*    sS  = (float*)smem;                      // [64][1024]
    _Float16* sQ  = (_Float16*)(smem + SQ_OFF);        // [64][64]
    _Float16* sKV = (_Float16*)(smem + SKV_OFF);       // [64][64]

    const int tid  = threadIdx.x;
    const int wid  = tid >> 5;
    const int lane = tid & 31;
    const int q0 = blockIdx.x * 64;
    const int h  = blockIdx.y;
    const int b  = blockIdx.z;
    const float scale = 0.14433756729740643f;   // 1/sqrt(48)

    // ---- stage Q once (async global->LDS) ----
    stage_rows48_async(Qg + (size_t)(b * SS + q0) * DD + h * PDIM, SQ_OFF, sQ, tid);
    wait_async0();

    // ---------------- Phase 1: scores = Q K^T ----------------
    const int t0  = wid * 2;
    const int trS = t0 >> 2;
    const int tcA = t0 & 3, tcB = (t0 + 1) & 3;

    for (int j = 0; j < 16; ++j) {
        __syncthreads();
        // K block (async): sKV[key][pd] == Bt[n=key][k=pd]
        stage_rows48_async(Kg + (size_t)(b * SS + j * 64) * DD + h * PDIM, SKV_OFF, sKV, tid);
        wait_async0();
        __syncthreads();

        f32x8 acc0 = {}, acc1 = {};
#pragma unroll
        for (int ks = 0; ks < 2; ++ks) {   // pd 0..31, 32..63 (zero padded)
            f16x16 af  = load_frag_a_f16(sQ + (trS * 16) * 64 + ks * 32, 64, lane);
            f16x16 bf0 = load_frag_b(sKV + (tcA * 16) * 64 + ks * 32, 64, lane);
            f16x16 bf1 = load_frag_b(sKV + (tcB * 16) * 64 + ks * 32, 64, lane);
            acc0 = wmma_f16(af, bf0, acc0);
            acc1 = wmma_f16(af, bf1, acc1);
        }
        const int hf = lane >> 4, nl = lane & 15;
#pragma unroll
        for (int r = 0; r < 8; ++r) {
            int row = trS * 16 + r + hf * 8;
            sS[row * 1024 + j * 64 + tcA * 16 + nl] = acc0[r];
            sS[row * 1024 + j * 64 + tcB * 16 + nl] = acc1[r];
        }
    }
    __syncthreads();

    // ------------- softmax: 64 rows, 4 threads/row, f32x4 passes -------------
    {
        int r = tid >> 2;
        float* rp = sS + r * 1024 + (tid & 3) * 256;
        float mx = -1e30f;
        for (int c4 = 0; c4 < 64; ++c4) {
            f32x4 v = *(const f32x4*)(rp + c4 * 4);
            mx = fmaxf(mx, fmaxf(fmaxf(v[0], v[1]), fmaxf(v[2], v[3])));
        }
        mx = fmaxf(mx, __shfl_xor(mx, 1, 32));
        mx = fmaxf(mx, __shfl_xor(mx, 2, 32));
        float sum = 0.f;
        for (int c4 = 0; c4 < 64; ++c4) {
            f32x4 v = *(const f32x4*)(rp + c4 * 4);
#pragma unroll
            for (int i = 0; i < 4; ++i) {
                v[i] = __expf((v[i] - mx) * scale);
                sum += v[i];
            }
            *(f32x4*)(rp + c4 * 4) = v;
        }
        sum += __shfl_xor(sum, 1, 32);
        sum += __shfl_xor(sum, 2, 32);
        float inv = 1.0f / sum;
        for (int c4 = 0; c4 < 64; ++c4) {
            f32x4 v = *(const f32x4*)(rp + c4 * 4);
#pragma unroll
            for (int i = 0; i < 4; ++i) v[i] *= inv;
            *(f32x4*)(rp + c4 * 4) = v;
        }
    }
    __syncthreads();

    // ---------------- Phase 2: O = P V ----------------
    // 12 output tiles (4 q-rows x 3 pd-cols): wave w takes tile w,
    // waves 0..3 also take tile w+8 (uniform per wave -> EXEC stays full).
    const int tA = wid, tB = wid + 8;
    const int trA = tA / 3, tcA2 = tA % 3;
    const int trB = tB / 3, tcB2 = tB % 3;
    f32x8 accA = {}, accB = {};

    for (int j = 0; j < 16; ++j) {
        __syncthreads();
        // stage V transposed (manual scatter): sKV[pd][key] == Bt[n=pd][k=key]
        {
            const _Float16* vsrc = Vg + (size_t)(b * SS + j * 64) * DD + h * PDIM;
            {
                int key = tid / 6, ch = tid - key * 6;
                f16x8 v = *(const f16x8*)(vsrc + (size_t)key * DD + ch * 8);
#pragma unroll
                for (int i = 0; i < 8; ++i) sKV[(ch * 8 + i) * 64 + key] = v[i];
            }
            if (tid < 128) {
                int tk = tid + 256;
                int key = tk / 6, ch = tk - key * 6;
                f16x8 v = *(const f16x8*)(vsrc + (size_t)key * DD + ch * 8);
#pragma unroll
                for (int i = 0; i < 8; ++i) sKV[(ch * 8 + i) * 64 + key] = v[i];
            } else {
                int tk2 = tid - 128;               // zero pad pd rows 48..63
                int pdr = 48 + (tk2 >> 3), c8 = tk2 & 7;
                f16x8 z = {};
                *(f16x8*)(sKV + pdr * 64 + c8 * 8) = z;
            }
        }
        __syncthreads();

#pragma unroll
        for (int ks = 0; ks < 2; ++ks) {   // key 0..31, 32..63 of this block
            int kbase = j * 64 + ks * 32;
            f16x16 a0 = load_frag_a_f32(sS + (trA * 16) * 1024 + kbase, 1024, lane);
            f16x16 b0 = load_frag_b(sKV + (tcA2 * 16) * 64 + ks * 32, 64, lane);
            accA = wmma_f16(a0, b0, accA);
            if (wid < 4) {
                f16x16 a1 = load_frag_a_f32(sS + (trB * 16) * 1024 + kbase, 1024, lane);
                f16x16 b1 = load_frag_b(sKV + (tcB2 * 16) * 64 + ks * 32, 64, lane);
                accB = wmma_f16(a1, b1, accB);
            }
        }
    }

    const int hf = lane >> 4, nl = lane & 15;
#pragma unroll
    for (int r = 0; r < 8; ++r) {
        int row = q0 + trA * 16 + r + hf * 8;
        Og[(size_t)(b * SS + row) * DD + h * PDIM + tcA2 * 16 + nl] = (_Float16)accA[r];
    }
    if (wid < 4) {
#pragma unroll
        for (int r = 0; r < 8; ++r) {
            int row = q0 + trB * 16 + r + hf * 8;
            Og[(size_t)(b * SS + row) * DD + h * PDIM + tcB2 * 16 + nl] = (_Float16)accB[r];
        }
    }
}

// =====================================================================
// Host launcher
// =====================================================================
extern "C" void kernel_launch(void* const* d_in, const int* in_sizes, int n_in,
                              void* d_out, int out_size, void* d_ws, size_t ws_size,
                              hipStream_t stream) {
    const float* X  = (const float*)d_in[0];
    const float* Wq = (const float*)d_in[1];
    const float* bq = (const float*)d_in[2];
    const float* Wk = (const float*)d_in[3];
    const float* bk = (const float*)d_in[4];
    const float* Wv = (const float*)d_in[5];
    const float* bv = (const float*)d_in[6];
    const float* Wo = (const float*)d_in[7];
    const float* bo = (const float*)d_in[8];
    float* out = (float*)d_out;

    // workspace: Q, K, V, concat  (each 8192x768 f16 = 12.6 MB)
    const size_t elems = (size_t)MTOT * DD;
    _Float16* qb = (_Float16*)d_ws;
    _Float16* kb = qb + elems;
    _Float16* vb = kb + elems;
    _Float16* cb = vb + elems;

    dim3 gGemm(DD / 128, MTOT / 128);   // (6, 64)
    dim3 blk(256);

    // QKV projections: f32 in -> f16 out
    gemm_wmma_kernel<float, _Float16><<<gGemm, blk, 0, stream>>>(X, Wq, bq, qb, MTOT, DD, DD);
    gemm_wmma_kernel<float, _Float16><<<gGemm, blk, 0, stream>>>(X, Wk, bk, kb, MTOT, DD, DD);
    gemm_wmma_kernel<float, _Float16><<<gGemm, blk, 0, stream>>>(X, Wv, bv, vb, MTOT, DD, DD);

    // Attention (272 KB dynamic LDS per workgroup)
    (void)hipFuncSetAttribute((const void*)attn_wmma_kernel,
                              hipFuncAttributeMaxDynamicSharedMemorySize,
                              ATTN_LDS_BYTES);
    dim3 gAttn(SS / 64, HH, BB);        // (16, 16, 8)
    attn_wmma_kernel<<<gAttn, blk, ATTN_LDS_BYTES, stream>>>(qb, kb, vb, cb);

    // Output projection: f16 in -> f32 out (+bias)
    gemm_wmma_kernel<_Float16, float><<<gGemm, blk, 0, stream>>>(cb, Wo, bo, out, MTOT, DD, DD);
}
